// PosteriorOUEncoder_28003186769981
// MI455X (gfx1250) — matrix operand order repaired and checked by reading, not compiled, not run on hardware
//
#include <hip/hip_runtime.h>
#include <hip/hip_bf16.h>
#include <math.h>

typedef __attribute__((ext_vector_type(2)))  float  v2f;
typedef __attribute__((ext_vector_type(8)))  float  v8f;
typedef __attribute__((ext_vector_type(16))) __bf16 v16bf;

#define NB   8
#define TT   2048
#define TM1  2047
#define XD   64
#define HD   256
#define CD   32
#define GD   768   // 3*H

static __device__ inline v8f wmma4(v2f a, v2f b, v8f c) {
  // V_WMMA_F32_16X16X4_F32  (D = A[16x4] * B[4x16] + C)
  return __builtin_amdgcn_wmma_f32_16x16x4_f32(false, a, false, b, (short)0, c, false, false);
}
static __device__ inline v8f wmma_bf16(v16bf a, v16bf b, v8f c) {
  // V_WMMA_F32_16X16X32_BF16
  return __builtin_amdgcn_wmma_f32_16x16x32_bf16(false, a, false, b, (short)0, c, false, false);
}

static __device__ inline float sigmoidf_(float x) { return 1.0f / (1.0f + expf(-x)); }
static __device__ inline float siluf_(float x)    { return x / (1.0f + expf(-x)); }
static __device__ inline float softplusf_(float x){ return (x > 20.0f) ? x : log1pf(expf(x)); }

// ---------------------------------------------------------------------------
// K1: gi = x @ W_ih^T + b_ih      [B*T,64] x [64,768]
// one 16x16 output tile per wave, fp32 WMMA (K=4), 8 waves/block
// ---------------------------------------------------------------------------
__global__ __launch_bounds__(256) void k1_gi_gemm(const float* __restrict__ x,
                                                  const float* __restrict__ Wih,
                                                  const float* __restrict__ bih,
                                                  float* __restrict__ gi) {
  const int tid  = threadIdx.x;
  const int lane = tid & 31, wv = tid >> 5;
  const int m = lane & 15, hi = lane >> 4, col = lane & 15;
  const int w  = blockIdx.x * 8 + wv;          // 0..49151
  const int mt = w / 48, nt = w % 48;
  const int row = mt * 16 + m;
  const int n   = nt * 16 + col;

  v8f acc = {0.f,0.f,0.f,0.f,0.f,0.f,0.f,0.f};
  for (int k0 = 0; k0 < XD; k0 += 4) {
    const int kk = k0 + 2 * hi;
    v2f av = *(const v2f*)&x[(long)row * XD + kk];
    v2f bv;
    bv.x = Wih[n * XD + kk];
    bv.y = Wih[n * XD + kk + 1];
    acc = wmma4(av, bv, acc);
  }
  const float bias = bih[n];
#pragma unroll
  for (int r = 0; r < 8; ++r) {
    const int rr = mt * 16 + r + 8 * hi;
    gi[(long)rr * GD + n] = acc[r] + bias;
  }
}

// ---------------------------------------------------------------------------
// K2: GRU sequential scan.  Single workgroup (512 threads = 16 waves).
// h: [16,H] in LDS (rows 8..15 zero padding).  Per step:
//   gh = h @ W_hh^T + b_hh   via fp32 WMMA (16 waves x 3 N-tiles)
//   gate update (VALU), h_seq store.
// ---------------------------------------------------------------------------
__global__ __launch_bounds__(512) void k2_gru_scan(const float* __restrict__ gi,
                                                   const float* __restrict__ Whh,
                                                   const float* __restrict__ bhh,
                                                   float* __restrict__ hseq,
                                                   float* __restrict__ hglob) {
  __shared__ float hsh[16][258];   // padded pitch (bank spread)
  __shared__ float ghs[8][770];

  const int tid  = threadIdx.x;
  const int lane = tid & 31, wv = tid >> 5;
  const int m = lane & 15, hi = lane >> 4, col = lane & 15;

  for (int i = tid; i < 16 * 258; i += 512) ((float*)hsh)[i] = 0.0f;
  __syncthreads();

#pragma unroll 1
  for (int step = 0; step < TT; ++step) {
    // ---- gh = h @ W_hh^T ----
#pragma unroll 1
    for (int tix = 0; tix < 3; ++tix) {
      const int nt = wv + tix * 16;          // 0..47
      const int n  = nt * 16 + col;
      v8f acc = {0.f,0.f,0.f,0.f,0.f,0.f,0.f,0.f};
      for (int k0 = 0; k0 < HD; k0 += 4) {
        const int kk = k0 + 2 * hi;
        v2f av = *(const v2f*)&hsh[m][kk];
        v2f bv;
        bv.x = Whh[n * HD + kk];
        bv.y = Whh[n * HD + kk + 1];
        acc = wmma4(av, bv, acc);
      }
      const float bias = bhh[n];
      if (hi == 0) {                          // rows 0..7 are the real batches
#pragma unroll
        for (int r = 0; r < 8; ++r) ghs[r][n] = acc[r] + bias;
      }
    }
    __syncthreads();
    // ---- gates ----
#pragma unroll
    for (int q = 0; q < 4; ++q) {
      const int idx = tid * 4 + q;            // 0..2047
      const int b = idx >> 8, hh = idx & 255;
      const long gbase = (long)(b * TT + step) * GD;
      const float gir = gi[gbase + hh];
      const float giz = gi[gbase + 256 + hh];
      const float gin = gi[gbase + 512 + hh];
      const float r = sigmoidf_(gir + ghs[b][hh]);
      const float z = sigmoidf_(giz + ghs[b][256 + hh]);
      const float nn = tanhf(gin + r * ghs[b][512 + hh]);
      const float hp = hsh[b][hh];
      const float hnew = (1.0f - z) * nn + z * hp;
      hseq[(long)(b * TT + step) * HD + hh] = hnew;
      hsh[b][hh] = hnew;
    }
    __syncthreads();
  }
#pragma unroll
  for (int q = 0; q < 4; ++q) {
    const int idx = tid * 4 + q;
    hglob[idx] = hsh[idx >> 8][idx & 255];
  }
}

// ---------------------------------------------------------------------------
// K3: banded soft attention.  logits = -(2048*(tk-tj))^2, max logit == 0, so
// unnormalized accumulation + row-sum normalization; window of 128 j's always
// contains the diagonal; contributions outside are < exp(-1000).
// One k-tile per block; 16 waves = 16 n-tiles; bf16 WMMA (K=32) x 4 chunks.
// ---------------------------------------------------------------------------
__global__ __launch_bounds__(512) void k3_attn(const float* __restrict__ t,
                                               const float* __restrict__ hseq,
                                               float* __restrict__ attn) {
  __shared__ float wsh[16][130];
  __shared__ float rsum[16];

  const int tid  = threadIdx.x;
  const int lane = tid & 31, wv = tid >> 5;
  const int m = lane & 15, hi = lane >> 4, col = lane & 15;

  const int bt = blockIdx.x;            // 0..1023
  const int b  = bt >> 7;
  const int kt = bt & 127;
  const int k0row = kt * 16;
  int jt0 = kt - 4;
  if (jt0 < 0) jt0 = 0;
  if (jt0 > 120) jt0 = 120;
  const int j0 = jt0 * 16;

  for (int e = tid; e < 16 * 128; e += 512) {
    const int kr = e >> 7, jj = e & 127;
    const float tk = t[b * TT + k0row + kr];
    const float tj = t[b * TT + j0 + jj];
    const float d  = 2048.0f * (tk - tj);
    wsh[kr][jj] = expf(-d * d);
  }
  __syncthreads();
  if (tid < 16) {
    float s = 0.0f;
    for (int jj = 0; jj < 128; ++jj) s += wsh[tid][jj];
    rsum[tid] = s;
  }
  __syncthreads();

  const int nt = wv;
  const int n  = nt * 16 + col;
  v8f acc = {0.f,0.f,0.f,0.f,0.f,0.f,0.f,0.f};
  for (int cch = 0; cch < 4; ++cch) {
    const int kb = cch * 32;
    v16bf a, bb;
#pragma unroll
    for (int v = 0; v < 8; ++v) {
      const int base16 = (v < 4) ? 0 : 16;
      const int vv = v & 3;
      const int kk = kb + base16 + 2 * vv + 8 * hi;   // A: 16-bit 16x32 layout
      a[2 * v]     = (__bf16)wsh[m][kk];
      a[2 * v + 1] = (__bf16)wsh[m][kk + 1];
      const int j = j0 + kb + base16 + 2 * vv + 8 * hi;        // B: 16-bit 32x16 layout
      bb[2 * v]     = (__bf16)hseq[(long)(b * TT + j) * HD + n];
      bb[2 * v + 1] = (__bf16)hseq[(long)(b * TT + j + 1) * HD + n];
    }
    acc = wmma_bf16(a, bb, acc);
  }
#pragma unroll
  for (int r = 0; r < 8; ++r) {
    const int row = r + 8 * hi;
    const float inv = 1.0f / rsum[row];
    attn[(long)(b * TT + k0row + row) * HD + n] = acc[r] * inv;
  }
}

// ---------------------------------------------------------------------------
// K4: init head (tiny): hi = silu((hglob+attn0) @ Wi1^T + bi1);
//     m0/logs0 = hi @ Wi2^T + bi2;  s0 = softplus(logs0)+1e-6
// ---------------------------------------------------------------------------
__global__ __launch_bounds__(256) void k4_init_head(const float* __restrict__ hglob,
                                                    const float* __restrict__ attn,
                                                    const float* __restrict__ Wi1,
                                                    const float* __restrict__ bi1,
                                                    const float* __restrict__ Wi2,
                                                    const float* __restrict__ bi2,
                                                    float* __restrict__ m0s0) {
  __shared__ float hi_s[8][258];
  const int tid = threadIdx.x;     // 0..255 -> output index n
  for (int b = 0; b < NB; ++b) {
    float acc = bi1[tid];
    for (int k = 0; k < HD; ++k) {
      const float v = hglob[b * HD + k] + attn[(long)(b * TT) * HD + k];
      acc += v * Wi1[tid * HD + k];
    }
    hi_s[b][tid] = siluf_(acc);
  }
  __syncthreads();
  for (int o = tid; o < NB * 2 * CD; o += 256) {
    const int b = o >> 6, j = o & 63;
    float acc = bi2[j];
    for (int k = 0; k < HD; ++k) acc += hi_s[b][k] * Wi2[j * HD + k];
    m0s0[b * 64 + j] = (j < CD) ? acc : (softplusf_(acc) + 1e-6f);
  }
}

// ---------------------------------------------------------------------------
// K5: step head, fully fused per 16-row tile (one wave per block).
//  ctx = [hglob+attn, t];  hs1 = silu(ctx@Ws1^T+bs1);  hs2 = silu(hs1@Ws2^T+bs2)
//  params = hs2@Ws3^T+bs3.  K=257 handled as K=256 WMMA + rank-1 t column.
// ---------------------------------------------------------------------------
__global__ __launch_bounds__(32) void k5_step_head(const float* __restrict__ hglob,
                                                   const float* __restrict__ attn,
                                                   const float* __restrict__ t,
                                                   const float* __restrict__ Ws1,
                                                   const float* __restrict__ bs1,
                                                   const float* __restrict__ Ws2,
                                                   const float* __restrict__ bs2,
                                                   const float* __restrict__ Ws3,
                                                   const float* __restrict__ bs3,
                                                   float* __restrict__ params) {
  __shared__ float bufA[16][258];
  __shared__ float bufB[16][258];
  __shared__ float tcol[16];

  const int lane = threadIdx.x;
  const int m = lane & 15, hi = lane >> 4, col = lane & 15;
  const int wid = blockIdx.x;          // 0..1023
  const int b   = wid >> 7;
  const int tt0 = (wid & 127) * 16;

  for (int e = lane; e < 16 * 256; e += 32) {
    const int rr = e >> 8, kk = e & 255;
    const int tt = tt0 + rr;
    float v = 0.0f;
    if (tt < TM1) v = hglob[b * HD + kk] + attn[(long)(b * TT + tt) * HD + kk];
    bufA[rr][kk] = v;
  }
  if (lane < 16) {
    const int tt = tt0 + lane;
    tcol[lane] = (tt < TM1) ? t[b * TT + tt] : 0.0f;
  }
  __syncthreads();

  // ---- stage 1: bufB = silu(bufA @ Ws1[:, :256]^T + t*Ws1[:,256] + bs1) ----
#pragma unroll 1
  for (int nt = 0; nt < 16; ++nt) {
    const int n = nt * 16 + col;
    v8f acc = {0.f,0.f,0.f,0.f,0.f,0.f,0.f,0.f};
    for (int k0 = 0; k0 < HD; k0 += 4) {
      const int kk = k0 + 2 * hi;
      v2f av = *(const v2f*)&bufA[m][kk];
      v2f bv;
      bv.x = Ws1[n * 257 + kk];
      bv.y = Ws1[n * 257 + kk + 1];
      acc = wmma4(av, bv, acc);
    }
    const float wlast = Ws1[n * 257 + 256];
    const float bias  = bs1[n];
#pragma unroll
    for (int r = 0; r < 8; ++r) {
      const int row = r + 8 * hi;
      bufB[row][nt * 16 + col] = siluf_(acc[r] + tcol[row] * wlast + bias);
    }
  }
  __syncthreads();

  // ---- stage 2: bufA = silu(bufB @ Ws2^T + bs2) ----
#pragma unroll 1
  for (int nt = 0; nt < 16; ++nt) {
    const int n = nt * 16 + col;
    v8f acc = {0.f,0.f,0.f,0.f,0.f,0.f,0.f,0.f};
    for (int k0 = 0; k0 < HD; k0 += 4) {
      const int kk = k0 + 2 * hi;
      v2f av = *(const v2f*)&bufB[m][kk];
      v2f bv;
      bv.x = Ws2[n * HD + kk];
      bv.y = Ws2[n * HD + kk + 1];
      acc = wmma4(av, bv, acc);
    }
    const float bias = bs2[n];
#pragma unroll
    for (int r = 0; r < 8; ++r) {
      const int row = r + 8 * hi;
      bufA[row][nt * 16 + col] = siluf_(acc[r] + bias);
    }
  }
  __syncthreads();

  // ---- stage 3: params = bufA @ Ws3^T + bs3  (N = 96) ----
#pragma unroll 1
  for (int nt = 0; nt < 6; ++nt) {
    const int n = nt * 16 + col;
    v8f acc = {0.f,0.f,0.f,0.f,0.f,0.f,0.f,0.f};
    for (int k0 = 0; k0 < HD; k0 += 4) {
      const int kk = k0 + 2 * hi;
      v2f av = *(const v2f*)&bufA[m][kk];
      v2f bv;
      bv.x = Ws3[n * HD + kk];
      bv.y = Ws3[n * HD + kk + 1];
      acc = wmma4(av, bv, acc);
    }
    const float bias = bs3[n];
#pragma unroll
    for (int r = 0; r < 8; ++r) {
      const int row = r + 8 * hi;
      const int tt  = tt0 + row;
      if (tt < TM1) params[(long)(b * TT + tt) * 96 + n] = acc[r] + bias;
    }
  }
}

// ---------------------------------------------------------------------------
// K6: OU posterior.  The cumprod/cumsum composition telescopes exactly to
// c_t = A_t*c_{t-1} + (1-A_t)*mu_t + eps*sqrt(var);  (c-mean)^2/var == eps^2.
// One thread per (batch, channel) chain; wave-level (=batch) log_q reduction.
// ---------------------------------------------------------------------------
__global__ __launch_bounds__(256) void k6_ou_scan(const float* __restrict__ params,
                                                  const float* __restrict__ t,
                                                  const float* __restrict__ eps0,
                                                  const float* __restrict__ eps,
                                                  const float* __restrict__ m0s0,
                                                  float* __restrict__ out) {
  __shared__ float red[256];
  const float LOG2PI = 1.8378770664093453f;
  const int tid = threadIdx.x;
  const int b = tid >> 5, ch = tid & 31;

  const float m0 = m0s0[b * 64 + ch];
  const float s0 = m0s0[b * 64 + 32 + ch];
  const float e0 = eps0[b * CD + ch];
  const float c0 = m0 + s0 * e0;
  const float s0c = fmaxf(s0, 1e-12f);
  const float dd = (c0 - m0) / s0c;
  float lq = -0.5f * (dd * dd + 2.0f * logf(s0c) + LOG2PI);
  out[(long)(b * TT) * CD + ch] = c0;

  float cp = c0;
#pragma unroll 1
  for (int tt = 0; tt < TM1; ++tt) {
    const long pbase = (long)(b * TT + tt) * 96;
    const float mu  = params[pbase + ch];
    const float kap = softplusf_(params[pbase + 32 + ch]) + 1e-6f;
    const float sig = softplusf_(params[pbase + 64 + ch]) + 1e-6f;
    const float dt  = fmaxf(t[b * TT + tt + 1] - t[b * TT + tt], 1e-6f);
    const float A   = expf(-kap * dt);
    const float tkd = 2.0f * kap * dt;
    const float s2  = sig * sig;
    float Q;
    if (tkd < 1e-6f) Q = s2 * dt * (1.0f - kap * dt + tkd * tkd * (1.0f / 6.0f));
    else             Q = s2 * (1.0f - expf(-tkd)) / fmaxf(2.0f * kap, 1e-12f);
    const float var = fmaxf(Q, 1e-12f);
    const float ep  = eps[(long)(b * TM1 + tt) * CD + ch];
    const float noise = ep * sqrtf(var);
    const float cn = A * cp + (1.0f - A) * mu + noise;
    out[(long)(b * TT + tt + 1) * CD + ch] = cn;
    lq += -0.5f * (ep * ep + logf(var) + LOG2PI);
    cp = cn;
  }
  red[tid] = lq;
  __syncthreads();
  if (ch == 0) {
    float s = 0.0f;
    for (int i = 0; i < 32; ++i) s += red[b * 32 + i];
    out[(long)NB * TT * CD + b] = s;
  }
}

// ---------------------------------------------------------------------------
extern "C" void kernel_launch(void* const* d_in, const int* in_sizes, int n_in,
                              void* d_out, int out_size, void* d_ws, size_t ws_size,
                              hipStream_t stream) {
  const float* x    = (const float*)d_in[0];
  const float* t    = (const float*)d_in[1];
  const float* eps0 = (const float*)d_in[2];
  const float* eps  = (const float*)d_in[3];
  const float* Wih  = (const float*)d_in[4];
  const float* Whh  = (const float*)d_in[5];
  const float* bih  = (const float*)d_in[6];
  const float* bhh  = (const float*)d_in[7];
  const float* Wi1  = (const float*)d_in[8];
  const float* bi1  = (const float*)d_in[9];
  const float* Wi2  = (const float*)d_in[10];
  const float* bi2  = (const float*)d_in[11];
  const float* Ws1  = (const float*)d_in[12];
  const float* bs1  = (const float*)d_in[13];
  const float* Ws2  = (const float*)d_in[14];
  const float* bs2  = (const float*)d_in[15];
  const float* Ws3  = (const float*)d_in[16];
  const float* bs3  = (const float*)d_in[17];

  float* wsf    = (float*)d_ws;
  float* gi     = wsf;                                    // B*T*3H  (50.3 MB)
  float* attnb  = wsf;                                    // overlay: gi dead after K2
  float* hseq   = wsf + (size_t)NB * TT * GD;             // B*T*H
  float* hglob  = hseq + (size_t)NB * TT * HD;            // B*H
  float* params = hglob + (size_t)NB * HD;                // B*T*3C (T-padded rows)
  float* m0s0   = params + (size_t)NB * TT * 3 * CD;      // B*64

  k1_gi_gemm<<<6144, 256, 0, stream>>>(x, Wih, bih, gi);
  k2_gru_scan<<<1, 512, 0, stream>>>(gi, Whh, bhh, hseq, hglob);
  k3_attn<<<1024, 512, 0, stream>>>(t, hseq, attnb);
  k4_init_head<<<1, 256, 0, stream>>>(hglob, attnb, Wi1, bi1, Wi2, bi2, m0s0);
  k5_step_head<<<1024, 32, 0, stream>>>(hglob, attnb, t, Ws1, bs1, Ws2, bs2, Ws3, bs3, params);
  k6_ou_scan<<<1, 256, 0, stream>>>(params, t, eps0, eps, m0s0, (float*)d_out);
}
